// SPL_15066745275147
// MI455X (gfx1250) — compile-verified, matrix operand order checked by reading, not run on hardware
//
#include <hip/hip_runtime.h>
#include <hip/hip_bf16.h>

// ---------------- model dims ----------------
#define SEEDLEN 120
#define TGTLEN  24
#define POSE    135
#define RNN     1024
#define LIN     256
#define SPLH    64
#define NSTEP   (SEEDLEN + TGTLEN - 1)  // 143
#define BATCH   256
#define TLEN    (SEEDLEN + TGTLEN)      // 144
#define K1PAD   1088                    // 1024 + 64 (parent region), mult of 32
#define XPAD    160                     // POSE=135 padded to mult of 32

typedef __attribute__((ext_vector_type(16))) __bf16 v16bf;
typedef __attribute__((ext_vector_type(8)))  float  v8f;

// joint DAG: number of parents and parent indices (by output order)
__constant__ int c_np[15] = {0,0,0,1,1,1,2,3,3,3,4,4,4,5,5};
__constant__ int c_par[15][5] = {
  {0,0,0,0,0},{0,0,0,0,0},{0,0,0,0,0},
  {0,0,0,0,0},{1,0,0,0,0},{2,0,0,0,0},
  {2,5,0,0,0},{2,5,6,0,0},{2,5,6,0,0},{2,5,6,0,0},
  {2,5,6,7,0},{2,5,6,8,0},{2,5,6,9,0},
  {2,5,6,8,11},{2,5,6,9,12}};

struct JointBias {
  const float* b1[15];
  const float* b2[15];
};

static __device__ __forceinline__ unsigned short f2bf(float f) {
  unsigned int u = __float_as_uint(f);
  unsigned int r = u + 0x7FFFu + ((u >> 16) & 1u);   // round-to-nearest-even
  return (unsigned short)(r >> 16);
}

union FragU { v16bf v; unsigned short u[16]; };

// Load a 16x32 bf16 fragment (A layout; B uses same layout with lane=col) from
// a row-major ushort matrix, rows contiguous along K. Per ISA 16-bit A layout:
// lanes 0-15 -> rows with K groups {0-7,16-23}, lanes 16-31 -> K {8-15,24-31}.
// Halves 0-7 and 8-15 are each K-contiguous -> two b128 loads, no repacking.
static __device__ __forceinline__ v16bf frag_ld_bf(const unsigned short* base, int ld) {
  int l = threadIdx.x & 31;
  const unsigned short* p = base + (l & 15) * ld + ((l >> 4) << 3);
  FragU r;
#pragma unroll
  for (int j = 0; j < 16; ++j) {
    int k = ((j >> 3) << 4) + (j & 7);
    r.u[j] = p[k];
  }
  return r.v;
}

static __device__ __forceinline__ v8f wmma_bf16(v16bf a, v16bf b, v8f c) {
  return __builtin_amdgcn_wmma_f32_16x16x32_bf16(false, a, false, b, (short)0, c, false, false);
}

static __device__ __forceinline__ float sigmoidf_(float x) {
  return 1.0f / (1.0f + __expf(-x));
}

// ---- async global->LDS staging (GLOBAL_LOAD_ASYNC_TO_LDS_B128, ASYNCcnt) ----
// Each lane: VDST = per-lane LDS byte address, VADDR = 64-bit global address;
// 16 bytes/lane/op flow memory->LDS without touching VGPRs (ISA 15.18.3 op 98).
static __device__ __forceinline__ void async_stage(const unsigned short* gsrc,
                                                   unsigned short* ldst,
                                                   int bytes, int nthreads) {
  unsigned lds_base = (unsigned)(size_t)ldst;          // low 32 bits = LDS addr
  unsigned long long gbase = (unsigned long long)(size_t)gsrc;
  for (int off = threadIdx.x * 16; off < bytes; off += nthreads * 16) {
    unsigned laddr = lds_base + (unsigned)off;
    unsigned long long ga = gbase + (unsigned long long)off;
    asm volatile("global_load_async_to_lds_b128 %0, %1, off"
                 :: "v"(laddr), "v"(ga) : "memory");
  }
}
static __device__ __forceinline__ void async_wait() {
  asm volatile("s_wait_asynccnt 0" ::: "memory");
}

// ---------------- one-time conversions ----------------
// src f32 [K][N] row-major -> dst bf16 [Npad][Kpad] (i.e. W^T, zero padded)
__global__ void k_cvt_t(const float* __restrict__ src, unsigned short* __restrict__ dst,
                        int K, int N, int Kpad, int Npad) {
  int idx = blockIdx.x * blockDim.x + threadIdx.x;
  int tot = Npad * Kpad;
  if (idx >= tot) return;
  int n = idx / Kpad, k = idx - n * Kpad;
  float v = (k < K && n < N) ? src[(long)k * N + n] : 0.0f;
  dst[idx] = f2bf(v);
}

// poses f32 [B][TLEN][POSE] -> x_bf bf16 [NSTEP][B][XPAD] (zero padded K)
__global__ void k_cvt_x(const float* __restrict__ poses, unsigned short* __restrict__ x_bf) {
  long idx = (long)blockIdx.x * blockDim.x + threadIdx.x;
  long tot = (long)NSTEP * BATCH * XPAD;
  if (idx >= tot) return;
  int k = (int)(idx % XPAD);
  long tm = idx / XPAD;
  int m = (int)(tm % BATCH);
  int t = (int)(tm / BATCH);
  float v = (k < POSE) ? poses[((long)m * TLEN + t) * POSE + k] : 0.0f;
  x_bf[idx] = f2bf(v);
}

__global__ void k_zero_state(unsigned short* __restrict__ hbf, float* __restrict__ c, int n) {
  int i = blockIdx.x * blockDim.x + threadIdx.x;
  if (i < n) { hbf[i] = 0; c[i] = 0.0f; }
}

// ---------------- s precompute: all timesteps in parallel ----------------
// s[t] = relu(x_t @ lin_W + lin_b), bf16 [NSTEP][256][256]. Pure bf16 WMMA.
__global__ void __launch_bounds__(32) k_s_all(const unsigned short* __restrict__ x_bf,
                                              const unsigned short* __restrict__ Wlin_t, // [256][160]
                                              const float* __restrict__ lin_b,
                                              unsigned short* __restrict__ s_all) {
  int tile = blockIdx.x & 255;
  int t    = blockIdx.x >> 8;
  int m0 = (tile >> 4) << 4;
  int n0 = (tile & 15) << 4;
  const unsigned short* A = x_bf + ((long)t * BATCH + m0) * XPAD;
  v8f acc = {};
#pragma unroll
  for (int kc = 0; kc < XPAD / 32; ++kc) {
    v16bf a = frag_ld_bf(A + kc * 32, XPAD);
    v16bf b = frag_ld_bf(Wlin_t + (long)n0 * XPAD + kc * 32, XPAD);
    acc = wmma_bf16(a, b, acc);
  }
  int l = threadIdx.x & 31;
  int n = n0 + (l & 15);
  int mbase = m0 + ((l >> 4) << 3);
  float bias = lin_b[n];
  unsigned short* s_t = s_all + (long)t * BATCH * LIN;
#pragma unroll
  for (int r = 0; r < 8; ++r) {
    float v = acc[r] + bias;
    v = v > 0.0f ? v : 0.0f;
    s_t[(long)(mbase + r) * LIN + n] = f2bf(v);
  }
}

// ---------------- recurrent step: LSTM cell ----------------
// Block = 4 waves, one 16-row m-tile x 64-col n-quad. The block async-stages
// the s[16x256] and h[16x1024] bf16 A-tiles into LDS (ASYNCcnt), then each
// wave computes a 16x16 n-tile for all 4 gates (A frags via ds_load, B frags
// via global b128), then applies the fused sigmoid/tanh cell update.
__global__ void __launch_bounds__(128) k_lstm(const unsigned short* __restrict__ s_t,
                                              const unsigned short* __restrict__ h_prev,
                                              const float* __restrict__ c_prev,
                                              const unsigned short* __restrict__ Wi_t, // [4096][256]
                                              const unsigned short* __restrict__ Wh_t, // [4096][1024]
                                              const float* __restrict__ bi,
                                              const float* __restrict__ bh,
                                              unsigned short* __restrict__ h_new,
                                              float* __restrict__ c_new) {
  __shared__ unsigned short sh_s[16 * LIN];   //  8 KB
  __shared__ unsigned short sh_h[16 * RNN];   // 32 KB
  const int tid  = threadIdx.x;
  const int wave = tid >> 5;
  int m0 = (blockIdx.x >> 4) << 4;
  int nq = (blockIdx.x & 15) << 6;

  // async stage A tiles (rows m0..m0+15 contiguous in both matrices)
  async_stage(s_t    + (long)m0 * LIN, sh_s, 16 * LIN * 2, 128);
  async_stage(h_prev + (long)m0 * RNN, sh_h, 16 * RNN * 2, 128);
  async_wait();
  __syncthreads();

  int n0 = nq + (wave << 4);
  v8f acc[4] = {{}, {}, {}, {}};
#pragma unroll
  for (int g = 0; g < 4; ++g) {
    const unsigned short* Bi = Wi_t + (long)(g * RNN + n0) * LIN;
    const unsigned short* Bh = Wh_t + (long)(g * RNN + n0) * RNN;
    __builtin_prefetch(Bh, 0, 3);            // global_prefetch_b8
    v8f a = acc[g];
    for (int kc = 0; kc < LIN / 32; ++kc) {
      v16bf av = frag_ld_bf(sh_s + kc * 32, LIN);
      v16bf bv = frag_ld_bf(Bi + kc * 32, LIN);
      a = wmma_bf16(av, bv, a);
    }
    for (int kc = 0; kc < RNN / 32; ++kc) {
      v16bf av = frag_ld_bf(sh_h + kc * 32, RNN);
      v16bf bv = frag_ld_bf(Bh + kc * 32, RNN);
      a = wmma_bf16(av, bv, a);
    }
    acc[g] = a;
  }

  int l = tid & 31;
  int n = n0 + (l & 15);
  int mbase = m0 + ((l >> 4) << 3);
  float bI = bi[n]           + bh[n];
  float bF = bi[RNN + n]     + bh[RNN + n];
  float bG = bi[2 * RNN + n] + bh[2 * RNN + n];
  float bO = bi[3 * RNN + n] + bh[3 * RNN + n];
#pragma unroll
  for (int r = 0; r < 8; ++r) {
    int m = mbase + r;
    float gi = sigmoidf_(acc[0][r] + bI);
    float gf = sigmoidf_(acc[1][r] + bF);
    float gg = tanhf(acc[2][r] + bG);
    float go = sigmoidf_(acc[3][r] + bO);
    float c  = gf * c_prev[(long)m * RNN + n] + gi * gg;
    float h  = go * tanhf(c);
    c_new[(long)m * RNN + n] = c;
    h_new[(long)m * RNN + n] = f2bf(h);
  }
}

// ---------------- joint MLP DAG + output ----------------
__global__ void __launch_bounds__(128) k_joints(const float* __restrict__ poses,
                                                const unsigned short* __restrict__ h_bf,
                                                const unsigned short* __restrict__ W1t, // [15][64][1088]
                                                const unsigned short* __restrict__ W2t, // [15][16][64]
                                                JointBias jb,
                                                float* __restrict__ out, int t) {
  __shared__ unsigned short sh_hh[16 * RNN];   // staged h tile, reused by all 15 joints
  __shared__ unsigned short sh_par[16 * 64];   // parent A-tile (K=1024..1087 region)
  __shared__ unsigned short sh_hid[16 * 64];   // layer-1 relu output, bf16
  __shared__ float sh_jout[15][16][9];         // joint outputs (f32)
  const int tid  = threadIdx.x;
  const int wave = tid >> 5;
  const int m0   = blockIdx.x << 4;

  // async stage h tile once per block
  async_stage(h_bf + (long)m0 * RNN, sh_hh, 16 * RNN * 2, 128);
  async_wait();
  __syncthreads();

  for (int j = 0; j < 15; ++j) {
    int np = c_np[j];
    for (int i = tid; i < 16 * 64; i += 128) {
      int m = i >> 6, c = i & 63;
      int pi = c / 9, q = c - pi * 9;
      float v = (pi < np) ? sh_jout[c_par[j][pi]][m][q] : 0.0f;
      sh_par[i] = f2bf(v);
    }
    __syncthreads();

    // layer 1: 64 output cols split as 4 waves x 16-col WMMA tiles
    {
      int n0 = wave << 4;
      const unsigned short* B = W1t + (long)j * 64 * K1PAD + (long)n0 * K1PAD;
      __builtin_prefetch(B, 0, 3);
      v8f acc = {};
      for (int kc = 0; kc < RNN / 32; ++kc) {        // h part, K=1024
        v16bf a = frag_ld_bf(sh_hh + kc * 32, RNN);
        v16bf b = frag_ld_bf(B + kc * 32, K1PAD);
        acc = wmma_bf16(a, b, acc);
      }
#pragma unroll
      for (int kc = 0; kc < 2; ++kc) {               // parent part, K=64
        v16bf a = frag_ld_bf(sh_par + kc * 32, 64);
        v16bf b = frag_ld_bf(B + 1024 + kc * 32, K1PAD);
        acc = wmma_bf16(a, b, acc);
      }
      int l = tid & 31;
      int n = n0 + (l & 15);
      int mb = (l >> 4) << 3;
      float bias = jb.b1[j][n];
#pragma unroll
      for (int r = 0; r < 8; ++r) {
        float v = acc[r] + bias;
        v = v > 0.0f ? v : 0.0f;
        sh_hid[(mb + r) * 64 + n] = f2bf(v);
      }
    }
    __syncthreads();

    // layer 2: [16x64] @ [64x9(pad16)] on wave 0
    if (wave == 0) {
      const unsigned short* B2 = W2t + (long)j * 16 * 64;
      v8f acc = {};
#pragma unroll
      for (int kc = 0; kc < 2; ++kc) {
        v16bf a = frag_ld_bf(sh_hid + kc * 32, 64);
        v16bf b = frag_ld_bf(B2 + kc * 32, 64);
        acc = wmma_bf16(a, b, acc);
      }
      int l = tid & 31;
      int n = l & 15;
      int mb = (l >> 4) << 3;
      if (n < 9) {
        float bias = jb.b2[j][n];
#pragma unroll
        for (int r = 0; r < 8; ++r)
          sh_jout[j][mb + r][n] = acc[r] + bias;
      }
    }
    __syncthreads();
  }

  // out = x_t + delta (only the last TGT timesteps are kept)
  if (t >= SEEDLEN - 1) {
    const float* x = poses + (long)m0 * (TLEN * POSE) + (long)t * POSE;
    int to = t - (SEEDLEN - 1);
    for (int i = tid; i < 16 * POSE; i += 128) {
      int m = i / POSE, c = i - m * POSE;
      int j = c / 9, q = c - j * 9;
      float v = x[(long)m * TLEN * POSE + c] + sh_jout[j][m][q];
      out[((long)(m0 + m) * TGTLEN + to) * POSE + c] = v;
    }
  }
}

// ---------------- host ----------------
extern "C" void kernel_launch(void* const* d_in, const int* in_sizes, int n_in,
                              void* d_out, int out_size, void* d_ws, size_t ws_size,
                              hipStream_t stream) {
  (void)in_sizes; (void)n_in; (void)out_size; (void)ws_size;
  const float* poses = (const float*)d_in[0];
  const float* lin_W = (const float*)d_in[1];
  const float* lin_b = (const float*)d_in[2];
  const float* Wi    = (const float*)d_in[3];
  const float* Wh    = (const float*)d_in[4];
  const float* bi    = (const float*)d_in[5];
  const float* bh    = (const float*)d_in[6];
  static const int npar[15] = {0,0,0,1,1,1,2,3,3,3,4,4,4,5,5};

  unsigned short* ws = (unsigned short*)d_ws;
  size_t off = 0;
  unsigned short* Wlin_t = ws + off; off += (size_t)256 * XPAD;
  unsigned short* Wi_t   = ws + off; off += (size_t)4096 * 256;
  unsigned short* Wh_t   = ws + off; off += (size_t)4096 * 1024;
  unsigned short* W1_t   = ws + off; off += (size_t)15 * 64 * K1PAD;
  unsigned short* W2_t   = ws + off; off += (size_t)15 * 16 * 64;
  unsigned short* x_bf   = ws + off; off += (size_t)NSTEP * BATCH * XPAD;
  unsigned short* s_all  = ws + off; off += (size_t)NSTEP * BATCH * LIN;
  unsigned short* h_bf   = ws + off; off += (size_t)2 * BATCH * RNN;
  off = (off + 1) & ~(size_t)1;
  float* c_st = (float*)(ws + off);   // 2 * 256 * 1024 floats

  auto cvt = [&](const float* src, unsigned short* dst, int K, int N, int Kpad, int Npad) {
    int tot = Npad * Kpad;
    k_cvt_t<<<(tot + 255) / 256, 256, 0, stream>>>(src, dst, K, N, Kpad, Npad);
  };
  cvt(lin_W, Wlin_t, POSE, LIN, XPAD, LIN);
  cvt(Wi,    Wi_t,   LIN,  4 * RNN, LIN, 4 * RNN);
  cvt(Wh,    Wh_t,   RNN,  4 * RNN, RNN, 4 * RNN);

  JointBias jb;
  for (int j = 0; j < 15; ++j) {
    const float* W1 = (const float*)d_in[7 + 4 * j + 0];
    jb.b1[j]        = (const float*)d_in[7 + 4 * j + 1];
    const float* W2 = (const float*)d_in[7 + 4 * j + 2];
    jb.b2[j]        = (const float*)d_in[7 + 4 * j + 3];
    int fin = RNN + 9 * npar[j];
    cvt(W1, W1_t + (size_t)j * 64 * K1PAD, fin, SPLH, K1PAD, SPLH);
    cvt(W2, W2_t + (size_t)j * 16 * 64,    SPLH, 9,    64,   16);
  }

  { // poses -> padded bf16, all timesteps
    long tot = (long)NSTEP * BATCH * XPAD;
    k_cvt_x<<<(int)((tot + 255) / 256), 256, 0, stream>>>(poses, x_bf);
  }

  k_zero_state<<<(BATCH * RNN + 255) / 256, 256, 0, stream>>>(h_bf, c_st, BATCH * RNN);

  // all 143 timesteps of s in one parallel launch (off the critical path)
  k_s_all<<<NSTEP * 256, 32, 0, stream>>>(x_bf, Wlin_t, lin_b, s_all);

  for (int t = 0; t < NSTEP; ++t) {
    size_t cur = (size_t)(t & 1) * BATCH * RNN;
    size_t nxt = (size_t)((t + 1) & 1) * BATCH * RNN;
    k_lstm<<<256, 128, 0, stream>>>(s_all + (size_t)t * BATCH * LIN,
                                    h_bf + cur, c_st + cur, Wi_t, Wh_t, bi, bh,
                                    h_bf + nxt, c_st + nxt);
    k_joints<<<16, 128, 0, stream>>>(poses, h_bf + nxt, W1_t, W2_t, jb, (float*)d_out, t);
  }
}
